// MultiScaleAttention_84473416777944
// MI455X (gfx1250) — compile-verified, matrix operand order checked by reading, not run on hardware
//
#include <hip/hip_runtime.h>
#include <hip/hip_bf16.h>

// ---------- types ----------
typedef __bf16 bf16_t;
typedef bf16_t v16bf __attribute__((ext_vector_type(16)));
typedef float  v8f  __attribute__((ext_vector_type(8)));

union FragA { v16bf v; unsigned int u[8]; };

__device__ __forceinline__ unsigned short f2bf(float f){
  unsigned int u = __float_as_uint(f);
  u += 0x7FFFu + ((u >> 16) & 1u);            // round-to-nearest-even
  return (unsigned short)(u >> 16);
}
__device__ __forceinline__ float bf2f(unsigned short s){
  return __uint_as_float(((unsigned int)s) << 16);
}

// ---------- CDNA5 async-to-LDS via inline asm (ISA: GLOBAL_LOAD_ASYNC_TO_LDS_B128) ----------
#if defined(__HIP_DEVICE_COMPILE__)
#define HAVE_ASYNC_LDS 1
#else
#define HAVE_ASYNC_LDS 0
#endif

#if HAVE_ASYNC_LDS
typedef __attribute__((address_space(3))) char lds_char_t;

// One 16-byte chunk per lane, memory -> LDS, tracked by ASYNCcnt.
__device__ __forceinline__ void async_cp16(const void* g, void* l){
  unsigned loff = (unsigned)(unsigned long long)(lds_char_t*)l;   // 32-bit LDS address
  asm volatile("global_load_async_to_lds_b128 %0, %1, off"
               :: "v"(loff), "v"((unsigned long long)g)
               : "memory");
}
__device__ __forceinline__ void wait_async0(){
#if __has_builtin(__builtin_amdgcn_s_wait_asynccnt)
  __builtin_amdgcn_s_wait_asynccnt(0);
#else
  asm volatile("s_wait_asynccnt 0" ::: "memory");
#endif
}
#endif

// 16x32 bf16 A-style fragment from row-major [16][>=32] matrix (ld in elems).
// ISA layout: lane L<16 -> row M=L, K={0..7} in V0-3, {16..23} in V4-7;
//             lane L+16 -> row M=L, K={8..15} in V0-3, {24..31} in V4-7.
__device__ __forceinline__ v16bf load_frag16x32(const unsigned short* p, int ld){
  int lane = threadIdx.x & 31;
  int half = lane >> 4;
  int r    = lane & 15;
  const unsigned short* row = p + (size_t)r * ld + 8 * half;
  FragA f;
#pragma unroll
  for (int v = 0; v < 4; ++v){
    f.u[v]     = *(const unsigned int*)(row + 2*v);
    f.u[v + 4] = *(const unsigned int*)(row + 16 + 2*v);
  }
  return f.v;
}

__device__ __forceinline__ v8f wmma_bf16(v16bf a, v16bf b, v8f c){
  return __builtin_amdgcn_wmma_f32_16x16x32_bf16(false, a, false, b, (short)0, c, false, false);
}

// ---------- problem constants ----------
#define BATCH 2
#define NTOK  32768          // 32^3
#define CDIM  384
#define C3    1152
#define NROWS (BATCH*NTOK)   // 65536
#define WS3   64
#define NG    512            // regions per batch (8^3)
#define LH    4              // local heads per scale
#define HD    32             // head dim

// ---------- conversion kernels ----------
__global__ void k_cvt_bf16(const float* __restrict__ in, unsigned short* __restrict__ out, int n){
  int i = blockIdx.x * blockDim.x + threadIdx.x;
  if (i < n) out[i] = f2bf(in[i]);
}

// in[rows][cols] f32 -> out[cols][rows] bf16 (store B transposed for WMMA B-fragments)
__global__ void k_transpose_bf16(const float* __restrict__ in, unsigned short* __restrict__ out,
                                 int rows, int cols){
  int i = blockIdx.x * blockDim.x + threadIdx.x;
  if (i >= rows * cols) return;
  int c = i % cols, r = i / cols;
  out[(size_t)c * rows + r] = f2bf(in[i]);
}

// bias[h][64][64] from rpb_table via relative-position index math
__global__ void k_bias(const float* __restrict__ rpb, float* __restrict__ bias){
  int t = blockIdx.x * blockDim.x + threadIdx.x;
  if (t >= LH * WS3 * WS3) return;
  int j  = t & 63;
  int ii = (t >> 6) & 63;
  int h  = t >> 12;
  int zi = ii >> 4, yi = (ii >> 2) & 3, xi = ii & 3;
  int zj = j  >> 4, yj = (j  >> 2) & 3, xj = j  & 3;
  int idx = (zi - zj + 3) * 11 + (yi - yj + 3) * 7 + (xi - xj + 3);
  bias[t] = rpb[idx * LH + h];
}

// ---------- WMMA GEMM: C[M][N] = A[M][K] * BT[N][K]^T ----------
// block = 256 threads (8 waves), wave tile 32x32 (2x2 WMMA), WG tile 64x128.
__global__ __launch_bounds__(256) void k_gemm_wmma(
    const unsigned short* __restrict__ A,
    const unsigned short* __restrict__ BT,
    int K, int N,
    unsigned short* __restrict__ outb,   // bf16 output (or null)
    float* __restrict__ outf,            // f32 output  (or null)
    const float* __restrict__ biasvec)   // per-col bias for f32 path (or null)
{
  const int lane = threadIdx.x & 31;
  const int wave = threadIdx.x >> 5;
  const int mbase = blockIdx.x * 64 + (wave & 1) * 32;
  const int nbase = blockIdx.y * 128 + (wave >> 1) * 32;

  v8f acc[2][2] = {};
  for (int kk = 0; kk < K; kk += 32){
    v16bf a0 = load_frag16x32(A  + (size_t)mbase        * K + kk, K);
    v16bf a1 = load_frag16x32(A  + (size_t)(mbase + 16) * K + kk, K);
    v16bf b0 = load_frag16x32(BT + (size_t)nbase        * K + kk, K);
    v16bf b1 = load_frag16x32(BT + (size_t)(nbase + 16) * K + kk, K);
    acc[0][0] = wmma_bf16(a0, b0, acc[0][0]);
    acc[0][1] = wmma_bf16(a0, b1, acc[0][1]);
    acc[1][0] = wmma_bf16(a1, b0, acc[1][0]);
    acc[1][1] = wmma_bf16(a1, b1, acc[1][1]);
  }
  const int radd = (lane >> 4) * 8;   // C layout: VGPR j holds rows j and 8+j
  const int cadd = lane & 15;
#pragma unroll
  for (int mi = 0; mi < 2; ++mi)
#pragma unroll
    for (int ni = 0; ni < 2; ++ni)
#pragma unroll
      for (int j = 0; j < 8; ++j){
        int row = mbase + mi * 16 + radd + j;
        int col = nbase + ni * 16 + cadd;
        float vv = acc[mi][ni][j];
        if (outb) outb[(size_t)row * N + col] = f2bf(vv);
        else      outf[(size_t)row * N + col] = vv + (biasvec ? biasvec[col] : 0.0f);
      }
}

// ---------- region pooling (mean + max over m^3 regions) ----------
// qkv: [NROWS][1152] bf16; pooled: [B][newR][64][3*128] bf16
__global__ void k_pool(const unsigned short* __restrict__ qkv,
                       unsigned short* __restrict__ pooled, int m, int scale_i){
  int g = 8 / m, newR = g * g * g;
  long total = (long)BATCH * newR * WS3 * 384;
  long t = (long)blockIdx.x * blockDim.x + threadIdx.x;
  if (t >= total) return;
  int c   = (int)(t % 384);
  int tok = (int)((t / 384) % WS3);
  int reg = (int)((t / (384L * WS3)) % newR);
  int b   = (int)( t / (384L * WS3 * newR));
  int p = c >> 7, ch = c & 127;
  int gx = reg % g, gy = (reg / g) % g, gz = reg / (g * g);
  int cbase = p * 384 + scale_i * 128 + ch;
  float inv = 1.0f / (float)(m * m * m);
  float sum = 0.f, mx = -3.4e38f;
  for (int mz = 0; mz < m; ++mz)
    for (int my = 0; my < m; ++my)
      for (int mq = 0; mq < m; ++mq){
        int sreg = (gz * m + mz) * 64 + (gy * m + my) * 8 + (gx * m + mq);
        size_t n = ((size_t)b * NG + sreg) * WS3 + tok;
        float v = bf2f(qkv[n * C3 + cbase]);
        sum += v; mx = fmaxf(mx, v);
      }
  pooled[t] = f2bf(sum * inv + mx);
}

// ---------- windowed attention (one block = one (b,reg,head), 4 waves) ----------
__global__ __launch_bounds__(128) void k_attn(
    const unsigned short* __restrict__ src, int ld, int pstride, int cbase, int R,
    const float* __restrict__ bias,
    unsigned short* __restrict__ fused,   // scale0 path (or null)
    float* __restrict__ yout,             // scale>0 path (or null)
    int ghead_base)
{
  __shared__ __align__(16) unsigned short lds_k[64 * 32];   // K rows x 32 ch
  __shared__ __align__(16) unsigned short lds_vt[32 * 64];  // V transposed [ch][tok]
  __shared__ __align__(16) unsigned short lds_p[4 * 16 * 64];

  const int tid  = threadIdx.x;
  const int lane = tid & 31;
  const int wave = tid >> 5;
  const int pb = blockIdx.x;
  const int h   = pb & 3;
  const int reg = (pb >> 2) % R;
  const int b   = pb / (4 * R);

  const size_t rowbase = ((size_t)(b * R + reg)) * WS3;
  const int qoff = cbase + h * HD;
  const int koff = qoff + pstride;
  const int voff = koff + pstride;

  const unsigned short* kb = src + rowbase * ld + koff;
  const unsigned short* vb = src + rowbase * ld + voff;

#if HAVE_ASYNC_LDS
  {
    // Async DMA staging: one b128 per lane per op (tracked by ASYNCcnt).
    // K -> lds_k (row-major, final layout). V rows -> lds_p (temp), transposed below.
    int row  = tid >> 2;              // 0..31
    int bofs = (tid & 3) * 16;        // byte chunk within the 64B row
    int lofs = tid * 16;              // dest byte offset (row*64 + chunk*16)
    async_cp16((const char*)(kb + (size_t)row        * ld) + bofs, (char*)lds_k + lofs);
    async_cp16((const char*)(kb + (size_t)(row + 32) * ld) + bofs, (char*)lds_k + 2048 + lofs);
    async_cp16((const char*)(vb + (size_t)row        * ld) + bofs, (char*)lds_p + lofs);
    async_cp16((const char*)(vb + (size_t)(row + 32) * ld) + bofs, (char*)lds_p + 2048 + lofs);
  }
  wait_async0();
  __syncthreads();
  // transpose temp V rows (lds_p) into lds_vt[ch][tok]
  for (int i = tid; i < 64 * 32; i += 128){
    int tk = i >> 5, c = i & 31;
    lds_vt[c * 64 + tk] = lds_p[i];
  }
  __syncthreads();
#else
  for (int i = tid; i < 64 * 32; i += 128){
    int tk = i >> 5, c = i & 31;
    lds_k[i]            = kb[(size_t)tk * ld + c];
    lds_vt[c * 64 + tk] = vb[(size_t)tk * ld + c];
  }
  __syncthreads();
#endif

  // S = scale * (q . k^T) + bias  ;  per wave: 16 query rows x 64 keys
  const unsigned short* qb = src + (rowbase + wave * 16) * ld + qoff;
  v16bf qf = load_frag16x32(qb, ld);
  v8f s[4];
#pragma unroll
  for (int t4 = 0; t4 < 4; ++t4){
    v16bf kf = load_frag16x32(lds_k + t4 * 16 * 32, 32);
    v8f z = {};
    s[t4] = wmma_bf16(qf, kf, z);
  }

  const float sc = 0.17677669529663687f;   // 32^-0.5
  const int radd = (lane >> 4) * 8;
  const int cadd = lane & 15;
  unsigned short* pw = lds_p + wave * 16 * 64;
#pragma unroll
  for (int j = 0; j < 8; ++j){
    int rowg = wave * 16 + radd + j;
    const float* brow = bias + ((size_t)h * WS3 + rowg) * WS3;
    float v0 = s[0][j] * sc + brow[cadd];
    float v1 = s[1][j] * sc + brow[16 + cadd];
    float v2 = s[2][j] * sc + brow[32 + cadd];
    float v3 = s[3][j] * sc + brow[48 + cadd];
    float mx = fmaxf(fmaxf(v0, v1), fmaxf(v2, v3));
#pragma unroll
    for (int o = 8; o >= 1; o >>= 1) mx = fmaxf(mx, __shfl_xor(mx, o, 32));
    v0 = __expf(v0 - mx); v1 = __expf(v1 - mx);
    v2 = __expf(v2 - mx); v3 = __expf(v3 - mx);
    float sm = v0 + v1 + v2 + v3;
#pragma unroll
    for (int o = 8; o >= 1; o >>= 1) sm += __shfl_xor(sm, o, 32);
    float rinv = 1.0f / sm;
    int rl = radd + j;
    pw[rl * 64 +      cadd] = f2bf(v0 * rinv);
    pw[rl * 64 + 16 + cadd] = f2bf(v1 * rinv);
    pw[rl * 64 + 32 + cadd] = f2bf(v2 * rinv);
    pw[rl * 64 + 48 + cadd] = f2bf(v3 * rinv);
  }

  // O = P (16x64) . V (64x32): 2 K-steps, 2 N-tiles
  v8f o0 = {}, o1 = {};
#pragma unroll
  for (int kk = 0; kk < 2; ++kk){
    v16bf pf  = load_frag16x32(pw + kk * 32, 64);
    v16bf vf0 = load_frag16x32(lds_vt + 0 * 16 * 64 + kk * 32, 64);
    v16bf vf1 = load_frag16x32(lds_vt + 1 * 16 * 64 + kk * 32, 64);
    o0 = wmma_bf16(pf, vf0, o0);
    o1 = wmma_bf16(pf, vf1, o1);
  }

  if (fused){
#pragma unroll
    for (int j = 0; j < 8; ++j){
      int tk = wave * 16 + radd + j;
      size_t n = rowbase + tk;
      size_t base = n * CDIM + (size_t)(ghead_base + h) * HD;
      fused[base + cadd]      = f2bf(o0[j]);
      fused[base + 16 + cadd] = f2bf(o1[j]);
    }
  } else {
#pragma unroll
    for (int j = 0; j < 8; ++j){
      int tk = wave * 16 + radd + j;
      size_t base = (((size_t)(b * LH + h) * R + reg) * WS3 + tk) * HD;
      yout[base + cadd]      = o0[j];
      yout[base + 16 + cadd] = o1[j];
    }
  }
}

// ---------- trilinear upsample of region grid r^3 -> 8^3, write into fused ----------
__global__ void k_upsample(const float* __restrict__ src, unsigned short* __restrict__ fused,
                           int r, int ghead_base){
  long t = (long)blockIdx.x * blockDim.x + threadIdx.x;
  if (t >= (1L << 23)) return;                  // 2*4*512*64*32
  int d   = (int)(t & 31);
  int tok = (int)((t >> 5) & 63);
  int reg = (int)((t >> 11) & 511);
  int h   = (int)((t >> 20) & 3);
  int b   = (int)(t >> 22);
  int R = r * r * r;
  int oz = reg >> 6, oy = (reg >> 3) & 7, ox = reg & 7;
  float fs = (float)r / 8.0f;
  float fz = fminf(fmaxf((oz + 0.5f) * fs - 0.5f, 0.f), (float)(r - 1));
  float fy = fminf(fmaxf((oy + 0.5f) * fs - 0.5f, 0.f), (float)(r - 1));
  float fx = fminf(fmaxf((ox + 0.5f) * fs - 0.5f, 0.f), (float)(r - 1));
  int z0 = (int)floorf(fz), y0 = (int)floorf(fy), x0 = (int)floorf(fx);
  int z1 = z0 + 1 < r ? z0 + 1 : r - 1;
  int y1 = y0 + 1 < r ? y0 + 1 : r - 1;
  int x1 = x0 + 1 < r ? x0 + 1 : r - 1;
  float wz = fz - z0, wy = fy - y0, wx = fx - x0;
  auto at = [&](int zz, int yy, int xx) -> float {
    int sreg = zz * r * r + yy * r + xx;
    return src[(((size_t)(b * LH + h) * R + sreg) * WS3 + tok) * HD + d];
  };
  float c00 = at(z0, y0, x0) * (1 - wx) + at(z0, y0, x1) * wx;
  float c01 = at(z0, y1, x0) * (1 - wx) + at(z0, y1, x1) * wx;
  float c10 = at(z1, y0, x0) * (1 - wx) + at(z1, y0, x1) * wx;
  float c11 = at(z1, y1, x0) * (1 - wx) + at(z1, y1, x1) * wx;
  float c0 = c00 * (1 - wy) + c01 * wy;
  float c1 = c10 * (1 - wy) + c11 * wy;
  float val = c0 * (1 - wz) + c1 * wz;
  fused[((size_t)(b * NG + reg) * WS3 + tok) * CDIM + (size_t)(ghead_base + h) * HD + d] = f2bf(val);
}

// ---------- host ----------
extern "C" void kernel_launch(void* const* d_in, const int* in_sizes, int n_in,
                              void* d_out, int out_size, void* d_ws, size_t ws_size,
                              hipStream_t stream) {
  const float* x      = (const float*)d_in[0];
  const float* w_qkv  = (const float*)d_in[1];
  const float* w_proj = (const float*)d_in[2];
  const float* b_proj = (const float*)d_in[3];
  const float* rpb    = (const float*)d_in[4];
  float* out = (float*)d_out;

  char* ws = (char*)d_ws;
  size_t off = 0;
  auto alloc = [&](size_t bytes) -> char* {
    off = (off + 255) & ~(size_t)255;
    char* p = ws + off;
    off += bytes;
    return p;
  };
  unsigned short* xb      = (unsigned short*)alloc((size_t)NROWS * CDIM * 2);
  unsigned short* wqkvT   = (unsigned short*)alloc((size_t)C3 * CDIM * 2);
  unsigned short* wprojT  = (unsigned short*)alloc((size_t)CDIM * CDIM * 2);
  unsigned short* qkv     = (unsigned short*)alloc((size_t)NROWS * C3 * 2);
  float*          bias    = (float*)alloc((size_t)LH * WS3 * WS3 * 4);
  unsigned short* pooled1 = (unsigned short*)alloc((size_t)BATCH * 64 * WS3 * 384 * 2);
  unsigned short* pooled2 = (unsigned short*)alloc((size_t)BATCH * 8  * WS3 * 384 * 2);
  float*          y1      = (float*)alloc((size_t)BATCH * LH * 64 * WS3 * HD * 4);
  float*          y2      = (float*)alloc((size_t)BATCH * LH * 8  * WS3 * HD * 4);
  unsigned short* fused   = (unsigned short*)alloc((size_t)NROWS * CDIM * 2);

  int n_x = NROWS * CDIM;
  k_cvt_bf16<<<(n_x + 255) / 256, 256, 0, stream>>>(x, xb, n_x);
  k_transpose_bf16<<<(CDIM * C3 + 255) / 256, 256, 0, stream>>>(w_qkv, wqkvT, CDIM, C3);
  k_transpose_bf16<<<(CDIM * CDIM + 255) / 256, 256, 0, stream>>>(w_proj, wprojT, CDIM, CDIM);
  k_bias<<<(LH * WS3 * WS3 + 255) / 256, 256, 0, stream>>>(rpb, bias);

  // qkv = x @ w_qkv  (bf16 in, f32 acc, bf16 out)
  k_gemm_wmma<<<dim3(NROWS / 64, C3 / 128), 256, 0, stream>>>(
      xb, wqkvT, CDIM, C3, qkv, nullptr, nullptr);

  // scale 0 attention -> fused channels [0,128)
  k_attn<<<BATCH * NG * LH, 128, 0, stream>>>(
      qkv, C3, CDIM, 0 * 128, NG, bias, fused, nullptr, 0);

  // pooled qkv for scales 1,2
  k_pool<<<((long)BATCH * 64 * WS3 * 384 + 255) / 256, 256, 0, stream>>>(qkv, pooled1, 2, 1);
  k_pool<<<((long)BATCH * 8  * WS3 * 384 + 255) / 256, 256, 0, stream>>>(qkv, pooled2, 4, 2);

  // scale 1/2 attention -> y buffers
  k_attn<<<BATCH * 64 * LH, 128, 0, stream>>>(
      pooled1, 384, 128, 0, 64, bias, nullptr, y1, 4);
  k_attn<<<BATCH * 8 * LH, 128, 0, stream>>>(
      pooled2, 384, 128, 0, 8, bias, nullptr, y2, 8);

  // upsample -> fused channels [128,256) and [256,384)
  k_upsample<<<((1L << 23) + 255) / 256, 256, 0, stream>>>(y1, fused, 4, 4);
  k_upsample<<<((1L << 23) + 255) / 256, 256, 0, stream>>>(y2, fused, 2, 8);

  // out = fused @ w_proj + b_proj (f32 out)
  k_gemm_wmma<<<dim3(NROWS / 64, CDIM / 128), 256, 0, stream>>>(
      fused, wprojT, CDIM, CDIM, nullptr, out, b_proj);
}
